// DGCM_33474975105266
// MI455X (gfx1250) — compile-verified, matrix operand order checked by reading, not run on hardware
//
#include <hip/hip_runtime.h>
#include <hip/hip_bf16.h>
#include <math.h>

// ---- problem constants (match reference) ----
#define B_     256
#define NWAYS  20
#define NSHOT  5
#define NKQ    175      // NT = N*K + Q
#define NKROWS 100      // N*K
#define Q_     75
#define D_     512
#define L_     4
#define KNEI   5
#define RPV    176      // padded rows per batch for V (11 tiles of 16)
#define RPP    32       // padded rows per batch for P (2 tiles of 16)
#define LQP    80       // padded query rows for logits (5 tiles of 16)
#define TAU_   0.1f
#define LSCALE 5.0f
#define ARES   0.2f

typedef __attribute__((ext_vector_type(16))) _Float16 v16h;
typedef __attribute__((ext_vector_type(8)))  _Float16 v8h;
typedef __attribute__((ext_vector_type(4)))  _Float16 v4h;
typedef __attribute__((ext_vector_type(8)))  float    v8f;
typedef int dgcm_v4i __attribute__((vector_size(16)));   // matches async builtin param

// CDNA5 async global->LDS path (ASYNCcnt), with safe fallback.
#if defined(__HIP_DEVICE_COMPILE__) && __has_builtin(__builtin_amdgcn_global_load_async_to_lds_b128)
#define DGCM_ASYNC 1
#else
#define DGCM_ASYNC 0
#endif

__device__ __forceinline__ void stage16(const _Float16* g, _Float16* l) {
#if DGCM_ASYNC
  __builtin_amdgcn_global_load_async_to_lds_b128(
      (__attribute__((address_space(1))) dgcm_v4i*)(unsigned long long)(const void*)g,
      (__attribute__((address_space(3))) dgcm_v4i*)(void*)l, 0, 0);
#else
  *(v8h*)l = *(const v8h*)g;
#endif
}

__device__ __forceinline__ void wait_async() {
#if DGCM_ASYNC
#if __has_builtin(__builtin_amdgcn_s_wait_asynccnt)
  __builtin_amdgcn_s_wait_asynccnt(0);
#else
  asm volatile("s_wait_asynccnt 0x0" ::: "memory");
#endif
#endif
}

// CDNA5 16-bit A/B fragment layout (ISA 7.12.2): lane = (m|n) + 16*g,
// per-lane 16 halves cover K = [8g..8g+7] and [16+8g..23+8g] of a 32-wide chunk.
__device__ __forceinline__ v16h load_frag16(const _Float16* rowk, int g) {
  const v8h lo = *(const v8h*)(rowk + 8 * g);
  const v8h hi = *(const v8h*)(rowk + 16 + 8 * g);
  v16h r;
#pragma unroll
  for (int e = 0; e < 8; ++e) { r[e] = lo[e]; r[e + 8] = hi[e]; }
  return r;
}

__device__ __forceinline__ float bred(float v, float* red) {
  const int t = threadIdx.x;
  red[t] = v;
  __syncthreads();
  for (int s = blockDim.x >> 1; s > 0; s >>= 1) {
    if (t < s) red[t] += red[t + s];
    __syncthreads();
  }
  float r = red[0];
  __syncthreads();
  return r;
}

// ---------------- init / utility kernels ----------------
__global__ void dgcm_zero_f32(float* p, int n) {
  int i = blockIdx.x * blockDim.x + threadIdx.x;
  if (i < n) p[i] = 0.0f;
}

__global__ void dgcm_zero_pad_rows(_Float16* buf, int rp, int rv) {
  const size_t per = (size_t)(rp - rv) * D_;
  const size_t total = (size_t)B_ * per;
  for (size_t i = (size_t)blockIdx.x * blockDim.x + threadIdx.x; i < total;
       i += (size_t)gridDim.x * blockDim.x) {
    size_t b = i / per, rem = i % per;
    size_t r = rv + rem / D_, d = rem % D_;
    buf[((size_t)b * rp + r) * D_ + d] = (_Float16)0.0f;
  }
}

__global__ void dgcm_f2h(const float* __restrict__ s, _Float16* __restrict__ d, size_t n) {
  for (size_t i = (size_t)blockIdx.x * blockDim.x + threadIdx.x; i < n;
       i += (size_t)gridDim.x * blockDim.x)
    d[i] = (_Float16)s[i];
}

// l2-normalize raw features into padded f32 + f32-copy + f16 mirror
__global__ __launch_bounds__(128)
void dgcm_init_norm(const float* __restrict__ in, float* __restrict__ X,
                    float* __restrict__ X0, _Float16* __restrict__ Xh,
                    int rp, int rv) {
  __shared__ float red[128];
  const int row = blockIdx.x;            // over B_*rv
  const int b = row / rv, r = row % rv;
  const int t = threadIdx.x;
  float4 v = ((const float4*)(in + ((size_t)b * rv + r) * D_))[t];
  float ss = v.x * v.x + v.y * v.y + v.z * v.z + v.w * v.w;
  ss = bred(ss, red);
  const float inv = 1.0f / fmaxf(sqrtf(ss), 1e-12f);
  v.x *= inv; v.y *= inv; v.z *= inv; v.w *= inv;
  const size_t o = ((size_t)b * rp + r) * D_;
  ((float4*)(X + o))[t] = v;
  ((float4*)(X0 + o))[t] = v;
  v4h h = {(_Float16)v.x, (_Float16)v.y, (_Float16)v.z, (_Float16)v.w};
  ((v4h*)(Xh + o))[t] = h;
}

// X = l2n((1-ARES)*T + ARES*X0), mirror to f16
__global__ __launch_bounds__(128)
void dgcm_resnorm(const float* __restrict__ T, const float* __restrict__ X0,
                  float* __restrict__ X, _Float16* __restrict__ Xh,
                  int rp, int rv) {
  __shared__ float red[128];
  const int row = blockIdx.x;
  const int b = row / rv, r = row % rv;
  const int t = threadIdx.x;
  const size_t o = ((size_t)b * rp + r) * D_;
  float4 a = ((const float4*)(T + o))[t];
  float4 z = ((const float4*)(X0 + o))[t];
  float4 v;
  v.x = (1.0f - ARES) * a.x + ARES * z.x;
  v.y = (1.0f - ARES) * a.y + ARES * z.y;
  v.z = (1.0f - ARES) * a.z + ARES * z.z;
  v.w = (1.0f - ARES) * a.w + ARES * z.w;
  float ss = bred(v.x * v.x + v.y * v.y + v.z * v.z + v.w * v.w, red);
  const float inv = 1.0f / fmaxf(sqrtf(ss), 1e-12f);
  v.x *= inv; v.y *= inv; v.z *= inv; v.w *= inv;
  ((float4*)(X + o))[t] = v;
  v4h h = {(_Float16)v.x, (_Float16)v.y, (_Float16)v.z, (_Float16)v.w};
  ((v4h*)(Xh + o))[t] = h;
}

// ---------------- generic single-wave NT WMMA GEMM ----------------
// C[b, m, n] = sum_k A[b, a_row0+m, k] * B[b, n, k]   (K = 512, f16 in, f32 out)
__global__ __launch_bounds__(32)
void dgcm_gemm_nt(const _Float16* __restrict__ A, int a_bs, int a_row0, int a_rp,
                  const _Float16* __restrict__ Bm, int b_bs,
                  float* __restrict__ C, int c_bs, int ldc,
                  int n_valid, float fill) {
  const int lane = threadIdx.x;
  const int g = lane >> 4, lr = lane & 15;
  const int b = blockIdx.z;
  int am = a_row0 + blockIdx.x * 16 + lr;
  if (am > a_rp - 1) am = a_rp - 1;                 // clamp into padded range
  const _Float16* Ar = A + ((size_t)b * a_bs + am) * D_;
  const int bn = blockIdx.y * 16 + lr;
  const _Float16* Br = Bm + ((size_t)b * b_bs + bn) * D_;
  __builtin_prefetch((const void*)Ar, 0, 0);
  __builtin_prefetch((const void*)Br, 0, 0);
  v8f acc = {};
  for (int k0 = 0; k0 < D_; k0 += 32) {
    v16h af = load_frag16(Ar + k0, g);
    v16h bf = load_frag16(Br + k0, g);
    acc = __builtin_amdgcn_wmma_f32_16x16x32_f16(false, af, false, bf,
                                                 (short)0, acc, false, false);
  }
  float* Cb = C + (size_t)b * c_bs;
  const int n = blockIdx.y * 16 + lr;
#pragma unroll
  for (int r = 0; r < 8; ++r) {
    const int m = blockIdx.x * 16 + r + 8 * g;      // D layout: M = r + 8*(lane>>4)
    Cb[(size_t)m * ldc + n] = (n < n_valid) ? acc[r] : fill;
  }
}

// ------- LDS double-buffered 8-wave GEMM (128x64 tile) + fused GNN epilogue -------
// out[m, n] = X[m, n] + s * tanh( (Ah · Wh^T)[m, n] + bias[n] )
// Each wave owns a 16-row stripe and all 64 columns: 1 A-frag + 4 B-frags + 4 WMMA
// per 32-wide K chunk; staging of chunk c+1 overlaps compute of chunk c (ASYNCcnt
// path when available).
__global__ __launch_bounds__(256)
void dgcm_gemm_upd(const _Float16* __restrict__ Ah, const _Float16* __restrict__ Wh,
                   const float* __restrict__ X, const float* __restrict__ bias,
                   const float* __restrict__ sptr, float* __restrict__ out) {
  __shared__ _Float16 As[2][128][48];
  __shared__ _Float16 Bs[2][64][48];
  const int t = threadIdx.x;
  const int w = t >> 5, lane = t & 31;              // 8 waves
  const int g = lane >> 4, lr = lane & 15;
  const size_t m0 = (size_t)blockIdx.x * 128;
  const int n0 = blockIdx.y * 64;

  // staging map: 16-byte (8-half) units
  const int ar0 = t >> 2, ac0 = (t & 3) * 8;        // A unit 0: rows 0..63
  const int ar1 = 64 + (t >> 2);                    // A unit 1: rows 64..127
  const int br = t >> 2, bc = (t & 3) * 8;          // B: 64 rows x 32 halves

  v8f acc[4] = {{}, {}, {}, {}};

  // prologue: stage chunk 0 into buffer 0
  stage16(Ah + (m0 + ar0) * D_ + ac0, &As[0][ar0][ac0]);
  stage16(Ah + (m0 + ar1) * D_ + ac0, &As[0][ar1][ac0]);
  stage16(Wh + (size_t)(n0 + br) * D_ + bc, &Bs[0][br][bc]);
  wait_async();
  __syncthreads();

  for (int c = 0; c < 16; ++c) {
    const int cur = c & 1;
    if (c < 15) {                                   // stage next chunk into other buffer
      const int nxt = cur ^ 1, k1 = (c + 1) * 32;
      stage16(Ah + (m0 + ar0) * D_ + k1 + ac0, &As[nxt][ar0][ac0]);
      stage16(Ah + (m0 + ar1) * D_ + k1 + ac0, &As[nxt][ar1][ac0]);
      stage16(Wh + (size_t)(n0 + br) * D_ + k1 + bc, &Bs[nxt][br][bc]);
    }
    v16h af = load_frag16(&As[cur][w * 16 + lr][0], g);
#pragma unroll
    for (int f = 0; f < 4; ++f) {
      v16h bf = load_frag16(&Bs[cur][f * 16 + lr][0], g);
      acc[f] = __builtin_amdgcn_wmma_f32_16x16x32_f16(false, af, false, bf,
                                                      (short)0, acc[f], false, false);
    }
    wait_async();
    __syncthreads();
  }

  const float s = *sptr;
#pragma unroll
  for (int f = 0; f < 4; ++f) {
    const int n = n0 + f * 16 + lr;
    const float bn = bias[n];
#pragma unroll
    for (int r = 0; r < 8; ++r) {
      const size_t m = m0 + w * 16 + r + 8 * g;
      const float x = X[m * D_ + n];
      out[m * D_ + n] = x + s * tanhf(acc[f][r] + bn);
    }
  }
}

// ---------------- top-k + softmax sparse adjacency ----------------
__global__ void dgcm_topk(const float* __restrict__ S, int ld, int rp, int rv, int cols,
                          int* __restrict__ idxO, float* __restrict__ wO) {
  const int row = blockIdx.x * blockDim.x + threadIdx.x;
  if (row >= B_ * rv) return;
  const int b = row / rv, r = row % rv;
  const float* s = S + ((size_t)b * rp + r) * ld;
  int idx[KNEI]; float val[KNEI];
#pragma unroll
  for (int j = 0; j < KNEI; ++j) {
    float best = -3.0e38f; int bi = 0;
    for (int c = 0; c < cols; ++c) {
      bool used = false;
      for (int u = 0; u < j; ++u) used = used || (idx[u] == c);
      const float x = s[c];
      if (!used && x > best) { best = x; bi = c; }
    }
    idx[j] = bi; val[j] = best;
  }
  const float mx = val[0];
  float w[KNEI], sum = 0.0f;
#pragma unroll
  for (int j = 0; j < KNEI; ++j) { w[j] = expf((val[j] - mx) / TAU_); sum += w[j]; }
  const float inv = 1.0f / sum;
  const size_t o = (size_t)row * KNEI;
#pragma unroll
  for (int j = 0; j < KNEI; ++j) { idxO[o + j] = idx[j]; wO[o + j] = w[j] * inv; }
}

// sparse message passing: Mh[row,:] = sum_j w_j * X[idx_j,:]  (f16 output for GEMM)
__global__ __launch_bounds__(128)
void dgcm_msg(const float* __restrict__ X, const int* __restrict__ idx,
              const float* __restrict__ w, _Float16* __restrict__ Mh,
              int rp, int rv) {
  const int row = blockIdx.x;
  const int b = row / rv, r = row % rv;
  const int t = threadIdx.x;
  const size_t base = (size_t)row * KNEI;
  float4 acc = {0.0f, 0.0f, 0.0f, 0.0f};
#pragma unroll
  for (int j = 0; j < KNEI; ++j) {
    const float wj = w[base + j];
    float4 v = ((const float4*)(X + ((size_t)b * rp + idx[base + j]) * D_))[t];
    acc.x += wj * v.x; acc.y += wj * v.y; acc.z += wj * v.z; acc.w += wj * v.w;
  }
  v4h h = {(_Float16)acc.x, (_Float16)acc.y, (_Float16)acc.z, (_Float16)acc.w};
  ((v4h*)(Mh + ((size_t)b * rp + r) * D_))[t] = h;
}

// VC = l2n(mean_shots V);  Pa = l2n(alpha*P + (1-alpha)*VC)  (f32 + f16 mirror)
__global__ __launch_bounds__(128)
void dgcm_vcpa(const float* __restrict__ V, const float* __restrict__ P,
               const float* __restrict__ alphap,
               float* __restrict__ Pa, _Float16* __restrict__ Pah) {
  __shared__ float red[128];
  const int b = blockIdx.x / NWAYS, c = blockIdx.x % NWAYS;
  const int t = threadIdx.x;
  float4 sum = {0.0f, 0.0f, 0.0f, 0.0f};
#pragma unroll
  for (int k = 0; k < NSHOT; ++k) {
    float4 v = ((const float4*)(V + ((size_t)b * RPV + c * NSHOT + k) * D_))[t];
    sum.x += v.x; sum.y += v.y; sum.z += v.z; sum.w += v.w;
  }
  const float inv5 = 1.0f / (float)NSHOT;
  sum.x *= inv5; sum.y *= inv5; sum.z *= inv5; sum.w *= inv5;
  float ss = bred(sum.x * sum.x + sum.y * sum.y + sum.z * sum.z + sum.w * sum.w, red);
  const float inv1 = 1.0f / fmaxf(sqrtf(ss), 1e-12f);
  sum.x *= inv1; sum.y *= inv1; sum.z *= inv1; sum.w *= inv1;
  const float alpha = *alphap;
  const size_t po = ((size_t)b * RPP + c) * D_;
  float4 p = ((const float4*)(P + po))[t];
  float4 pr;
  pr.x = alpha * p.x + (1.0f - alpha) * sum.x;
  pr.y = alpha * p.y + (1.0f - alpha) * sum.y;
  pr.z = alpha * p.z + (1.0f - alpha) * sum.z;
  pr.w = alpha * p.w + (1.0f - alpha) * sum.w;
  float ss2 = bred(pr.x * pr.x + pr.y * pr.y + pr.z * pr.z + pr.w * pr.w, red);
  const float inv2 = 1.0f / fmaxf(sqrtf(ss2), 1e-12f);
  pr.x *= inv2; pr.y *= inv2; pr.z *= inv2; pr.w *= inv2;
  ((float4*)(Pa + po))[t] = pr;
  v4h h = {(_Float16)pr.x, (_Float16)pr.y, (_Float16)pr.z, (_Float16)pr.w};
  ((v4h*)(Pah + po))[t] = h;
}

// softmax guidance: V = l2n(beta*V + (1-beta) * softmax(sims/tau) @ Pa)
__global__ __launch_bounds__(256)
void dgcm_guide(const float* __restrict__ sims, const float* __restrict__ Pa,
                const float* __restrict__ betap,
                float* __restrict__ V, _Float16* __restrict__ Vh) {
  __shared__ float w[NWAYS];
  __shared__ float red[256];
  const int row = blockIdx.x;
  const int b = row / NKQ, r = row % NKQ;
  const int t = threadIdx.x;
  if (t == 0) {
    const float* s = sims + ((size_t)b * RPV + r) * RPP;
    float tmp[NWAYS], mx = -3.0e38f;
    for (int m = 0; m < NWAYS; ++m) { tmp[m] = s[m]; mx = fmaxf(mx, tmp[m]); }
    float sum = 0.0f;
    for (int m = 0; m < NWAYS; ++m) { tmp[m] = expf((tmp[m] - mx) / TAU_); sum += tmp[m]; }
    const float inv = 1.0f / sum;
    for (int m = 0; m < NWAYS; ++m) w[m] = tmp[m] * inv;
  }
  __syncthreads();
  const size_t vb = ((size_t)b * RPV + r) * D_;
  const int d = t * 2;
  float g0 = 0.0f, g1 = 0.0f;
#pragma unroll 4
  for (int m = 0; m < NWAYS; ++m) {
    const float* pr = Pa + ((size_t)b * RPP + m) * D_ + d;
    const float wm = w[m];
    g0 += wm * pr[0]; g1 += wm * pr[1];
  }
  const float beta = *betap;
  float v0 = beta * V[vb + d] + (1.0f - beta) * g0;
  float v1 = beta * V[vb + d + 1] + (1.0f - beta) * g1;
  const float ss = bred(v0 * v0 + v1 * v1, red);
  const float inv = 1.0f / fmaxf(sqrtf(ss), 1e-12f);
  v0 *= inv; v1 *= inv;
  V[vb + d] = v0; V[vb + d + 1] = v1;
  Vh[vb + d] = (_Float16)v0; Vh[vb + d + 1] = (_Float16)v1;
}

// per-row cross-entropy (+ argmax accuracy on last layer)
__global__ void dgcm_ce(const float* __restrict__ logits, const int* __restrict__ y,
                        float* __restrict__ lacc, float* __restrict__ aacc, int do_acc) {
  const int i = blockIdx.x * blockDim.x + threadIdx.x;
  if (i >= B_ * Q_) return;
  const int b = i / Q_, q = i % Q_;
  const float* lr = logits + ((size_t)b * LQP + q) * RPP;
  const int yy = y[i];
  float l[NWAYS], mx = -3.0e38f; int am = 0;
#pragma unroll
  for (int n = 0; n < NWAYS; ++n) {
    l[n] = LSCALE * lr[n];
    if (l[n] > mx) { mx = l[n]; am = n; }
  }
  float sum = 0.0f;
#pragma unroll
  for (int n = 0; n < NWAYS; ++n) sum += expf(l[n] - mx);
  atomicAdd(lacc, mx + logf(sum) - l[yy]);
  if (do_acc) atomicAdd(aacc, (am == yy) ? 1.0f : 0.0f);
}

__global__ void dgcm_final(const float* __restrict__ lacc, const float* __restrict__ aacc,
                           float* __restrict__ out) {
  if (threadIdx.x == 0 && blockIdx.x == 0) {
    out[0] = lacc[0] / (float)(L_ * B_ * Q_);
    out[1] = aacc[0] / (float)(B_ * Q_);
  }
}

// ---------------- host orchestration ----------------
extern "C" void kernel_launch(void* const* d_in, const int* in_sizes, int n_in,
                              void* d_out, int out_size, void* d_ws, size_t ws_size,
                              hipStream_t stream) {
  (void)in_sizes; (void)n_in; (void)out_size; (void)ws_size;
  const float* V_feat = (const float*)d_in[0];
  const float* P_feat = (const float*)d_in[1];
  const int*   query_y = (const int*)d_in[2];
  const float* igb_W = (const float*)d_in[3];
  const float* igb_b = (const float*)d_in[4];
  const float* igb_s = (const float*)d_in[5];
  const float* pgb_W = (const float*)d_in[6];
  const float* pgb_b = (const float*)d_in[7];
  const float* pgb_s = (const float*)d_in[8];
  const float* alphap = (const float*)d_in[9];
  const float* betap  = (const float*)d_in[10];
  float* out = (float*)d_out;

  // bump allocator over workspace
  char* wsp = (char*)d_ws;
  size_t off = 0;
  auto alloc = [&](size_t bytes) -> char* {
    char* p = wsp + off;
    off += (bytes + 255) & ~(size_t)255;
    return p;
  };
  const size_t nVf = (size_t)B_ * RPV * D_;
  const size_t nPf = (size_t)B_ * RPP * D_;
  float*    V     = (float*)alloc(nVf * 4);
  float*    V0    = (float*)alloc(nVf * 4);
  float*    Vtmp  = (float*)alloc(nVf * 4);
  _Float16* Vh    = (_Float16*)alloc(nVf * 2);
  _Float16* msgVh = (_Float16*)alloc(nVf * 2);
  float*    P     = (float*)alloc(nPf * 4);
  float*    P0    = (float*)alloc(nPf * 4);
  float*    Ptmp  = (float*)alloc(nPf * 4);
  float*    Pa    = (float*)alloc(nPf * 4);
  _Float16* Ph    = (_Float16*)alloc(nPf * 2);
  _Float16* msgPh = (_Float16*)alloc(nPf * 2);
  _Float16* Pah   = (_Float16*)alloc(nPf * 2);
  _Float16* Wih   = (_Float16*)alloc((size_t)L_ * D_ * D_ * 2);
  _Float16* Wph   = (_Float16*)alloc((size_t)L_ * D_ * D_ * 2);
  float*    simI  = (float*)alloc((size_t)B_ * RPV * RPV * 4);
  float*    simP  = (float*)alloc((size_t)B_ * RPP * RPP * 4);
  float*    simsG = (float*)alloc((size_t)B_ * RPV * RPP * 4);
  float*    logits = (float*)alloc((size_t)B_ * LQP * RPP * 4);
  int*      idxI  = (int*)alloc((size_t)B_ * NKQ * KNEI * 4);
  float*    wI    = (float*)alloc((size_t)B_ * NKQ * KNEI * 4);
  int*      idxP  = (int*)alloc((size_t)B_ * NWAYS * KNEI * 4);
  float*    wP    = (float*)alloc((size_t)B_ * NWAYS * KNEI * 4);
  float*    accum = (float*)alloc(2 * 4);   // [0]=loss sum, [1]=correct count

  // init: accumulators, pad rows (workspace is poisoned), f16 weights, normalized feats
  dgcm_zero_f32<<<1, 64, 0, stream>>>(accum, 2);
  dgcm_zero_pad_rows<<<1024, 256, 0, stream>>>(Vh, RPV, NKQ);
  dgcm_zero_pad_rows<<<1024, 256, 0, stream>>>(msgVh, RPV, NKQ);
  dgcm_zero_pad_rows<<<1024, 256, 0, stream>>>(Ph, RPP, NWAYS);
  dgcm_zero_pad_rows<<<1024, 256, 0, stream>>>(msgPh, RPP, NWAYS);
  dgcm_zero_pad_rows<<<1024, 256, 0, stream>>>(Pah, RPP, NWAYS);
  dgcm_f2h<<<2048, 256, 0, stream>>>(igb_W, Wih, (size_t)L_ * D_ * D_);
  dgcm_f2h<<<2048, 256, 0, stream>>>(pgb_W, Wph, (size_t)L_ * D_ * D_);
  dgcm_init_norm<<<B_ * NKQ, 128, 0, stream>>>(V_feat, V, V0, Vh, RPV, NKQ);
  dgcm_init_norm<<<B_ * NWAYS, 128, 0, stream>>>(P_feat, P, P0, Ph, RPP, NWAYS);

  for (int i = 0; i < L_; ++i) {
    // ---- instance graph branch ----
    dgcm_gemm_nt<<<dim3(RPV / 16, RPV / 16, B_), 32, 0, stream>>>(
        Vh, RPV, 0, RPV, Vh, RPV, simI, RPV * RPV, RPV, NKQ, -1.0e30f);
    dgcm_topk<<<(B_ * NKQ + 255) / 256, 256, 0, stream>>>(
        simI, RPV, RPV, NKQ, NKQ, idxI, wI);
    dgcm_msg<<<B_ * NKQ, 128, 0, stream>>>(V, idxI, wI, msgVh, RPV, NKQ);
    dgcm_gemm_upd<<<dim3(B_ * RPV / 128, D_ / 64), 256, 0, stream>>>(
        msgVh, Wih + (size_t)i * D_ * D_, V, igb_b + (size_t)i * D_, igb_s + i, Vtmp);
    dgcm_resnorm<<<B_ * NKQ, 128, 0, stream>>>(Vtmp, V0, V, Vh, RPV, NKQ);

    // ---- prototype graph branch ----
    dgcm_gemm_nt<<<dim3(RPP / 16, RPP / 16, B_), 32, 0, stream>>>(
        Ph, RPP, 0, RPP, Ph, RPP, simP, RPP * RPP, RPP, NWAYS, -1.0e30f);
    dgcm_topk<<<(B_ * NWAYS + 255) / 256, 256, 0, stream>>>(
        simP, RPP, RPP, NWAYS, NWAYS, idxP, wP);
    dgcm_msg<<<B_ * NWAYS, 128, 0, stream>>>(P, idxP, wP, msgPh, RPP, NWAYS);
    dgcm_gemm_upd<<<dim3(B_ * RPP / 128, D_ / 64), 256, 0, stream>>>(
        msgPh, Wph + (size_t)i * D_ * D_, P, pgb_b + (size_t)i * D_, pgb_s + i, Ptmp);
    dgcm_resnorm<<<B_ * NWAYS, 128, 0, stream>>>(Ptmp, P0, P, Ph, RPP, NWAYS);

    // ---- adapted prototypes + guidance ----
    dgcm_vcpa<<<B_ * NWAYS, 128, 0, stream>>>(V, P, alphap, Pa, Pah);
    dgcm_gemm_nt<<<dim3(RPV / 16, RPP / 16, B_), 32, 0, stream>>>(
        Vh, RPV, 0, RPV, Pah, RPP, simsG, RPV * RPP, RPP, NWAYS, -1.0e30f);
    dgcm_guide<<<B_ * NKQ, 256, 0, stream>>>(simsG, Pa, betap, V, Vh);

    // ---- per-layer logits + CE ----
    dgcm_gemm_nt<<<dim3(LQP / 16, RPP / 16, B_), 32, 0, stream>>>(
        Vh, RPV, NKROWS, RPV, Pah, RPP, logits, LQP * RPP, RPP, NWAYS, 0.0f);
    dgcm_ce<<<(B_ * Q_ + 255) / 256, 256, 0, stream>>>(
        logits, query_y, accum, accum + 1, (i == L_ - 1) ? 1 : 0);
  }

  dgcm_final<<<1, 32, 0, stream>>>(accum, accum + 1, out);
}
